// AdaptDirGraphGonvLayer_49581102465772
// MI455X (gfx1250) — compile-verified, matrix operand order checked by reading, not run on hardware
//
#include <hip/hip_runtime.h>
#include <hip/hip_bf16.h>
#include <cstdint>

#define C_DIM 2048
#define KN 14
#define MB 112      // rows per GEMM block = 8 batches * 14 nodes
#define NB 128      // cols per GEMM block
#define KT 64       // K slice per LDS stage (2 WMMA k-steps)

typedef __attribute__((ext_vector_type(8)))  float  v8f;
typedef __attribute__((ext_vector_type(8)))  __bf16 v8bf;
typedef __attribute__((ext_vector_type(16))) __bf16 v16bf;

__device__ __forceinline__ uint32_t pack_bf16(float lo, float hi) {
    union { __bf16 h[2]; uint32_t u; } p;
    p.h[0] = (__bf16)lo;
    p.h[1] = (__bf16)hi;
    return p.u;
}

// ---------------- zero stats region ----------------
__global__ __launch_bounds__(256) void init_kernel(float* ws) {
    int i = blockIdx.x * 256 + threadIdx.x;
    if (i < 8192) ws[i] = 0.f;
}

// ---------------- BN stats: per-channel sum(gap), sum(gap^2) ----------------
__global__ __launch_bounds__(256)
void stats_kernel(const float* __restrict__ x, const int* __restrict__ ei,
                  const int* __restrict__ ej, int E,
                  float* __restrict__ Asum, float* __restrict__ Bsum) {
    __shared__ float dsh[13 * 256];
    __shared__ int eis[128], ejs[128];
    const int tid = threadIdx.x;
    const int b = blockIdx.x;
    if (tid < E) { eis[tid] = ei[tid]; ejs[tid] = ej[tid]; }
    __syncthreads();
    const float* xb = x + (size_t)b * KN * C_DIM;
    for (int cc = 0; cc < C_DIM; cc += 256) {
        int c = cc + tid;
        float g13 = xb[13 * C_DIM + c];
        #pragma unroll
        for (int i = 0; i < 13; ++i)
            dsh[i * 256 + tid] = fabsf(xb[i * C_DIM + c] - g13);
        float wsum = 0.f, p2 = 0.f;
        for (int e = 0; e < E; ++e) {
            float g = dsh[eis[e] * 256 + tid] - dsh[ejs[e] * 256 + tid];
            wsum += g; p2 += g * g;
        }
        atomicAdd(&Asum[c], 15.f * wsum);    // sum of gap
        atomicAdd(&Bsum[c], 225.f * p2);     // sum of gap^2
    }
}

// ---------------- coef[c] = gamma*inv*w_direct ; bias = sum (beta-mean*gamma*inv)*w_direct ----
__global__ __launch_bounds__(1024)
void coef_kernel(const float* __restrict__ Asum, const float* __restrict__ Bsum,
                 const float* __restrict__ gamma, const float* __restrict__ beta,
                 const float* __restrict__ wdir, int bs, int E,
                 float* __restrict__ coef, float* __restrict__ bias) {
    __shared__ float red[1024];
    const int tid = threadIdx.x;
    const float invN = 1.f / ((float)bs * (float)E);
    float t = 0.f;
    for (int c = tid; c < C_DIM; c += 1024) {
        float mean = Asum[c] * invN;
        float msq  = Bsum[c] * invN;
        float var  = msq - mean * mean;
        float inv  = rsqrtf(var + 1e-5f);
        float gi   = gamma[c] * inv;
        coef[c] = gi * wdir[c];
        t += (beta[c] - mean * gi) * wdir[c];
    }
    red[tid] = t;
    __syncthreads();
    for (int s = 512; s > 0; s >>= 1) {
        if (tid < s) red[tid] += red[tid + s];
        __syncthreads();
    }
    if (tid == 0) *bias = red[0];
}

// ---------------- per-batch: node dots S[i], edge sigmoid, L1-normalized adjacency ----
__global__ __launch_bounds__(256)
void adj_kernel(const float* __restrict__ x, const float* __restrict__ adjmask,
                const int* __restrict__ ei, const int* __restrict__ ej, int E,
                const float* __restrict__ coef, const float* __restrict__ bias,
                float* __restrict__ newadj) {
    __shared__ float red[256];
    __shared__ float S[14];
    __shared__ float am[196];
    __shared__ float rs[14];
    const int tid = threadIdx.x;
    const int b = blockIdx.x;
    const float* xb = x + (size_t)b * KN * C_DIM;
    float s[13];
    #pragma unroll
    for (int i = 0; i < 13; ++i) s[i] = 0.f;
    for (int c = tid; c < C_DIM; c += 256) {
        float cf = coef[c];
        float g13 = xb[13 * C_DIM + c];
        #pragma unroll
        for (int i = 0; i < 13; ++i)
            s[i] += fabsf(xb[i * C_DIM + c] - g13) * cf;
    }
    #pragma unroll
    for (int i = 0; i < 13; ++i) {
        red[tid] = s[i];
        __syncthreads();
        for (int st = 128; st > 0; st >>= 1) {
            if (tid < st) red[tid] += red[tid + st];
            __syncthreads();
        }
        if (tid == 0) S[i] = red[0];
        __syncthreads();
    }
    if (tid < 196) am[tid] = 1.f;
    __syncthreads();
    if (tid < E) {
        int i = ei[tid], j = ej[tid];
        float z = 3.f * (15.f * (S[i] - S[j]) + *bias);
        float a = 1.f / (1.f + __expf(-z));
        am[i * KN + j] = 2.f * a;
        am[j * KN + i] = 2.f * (1.f - a);
    }
    __syncthreads();
    if (tid < 196) am[tid] *= adjmask[tid];
    __syncthreads();
    if (tid < 14) {
        float sum = 0.f;
        #pragma unroll
        for (int j = 0; j < KN; ++j) sum += fabsf(am[tid * KN + j]);
        rs[tid] = fmaxf(sum, 1e-12f);
    }
    __syncthreads();
    if (tid < 196) newadj[(size_t)b * 196 + tid] = am[tid] / rs[tid / KN];
}

// ---------------- merged = newadj @ x, stored bf16 (dword-packed) ----------------
__global__ __launch_bounds__(256)
void merged_kernel(const float* __restrict__ x, const float* __restrict__ newadj,
                   uint32_t* __restrict__ mbf32) {
    __shared__ float A[196];
    const int tid = threadIdx.x, b = blockIdx.x;
    if (tid < 196) A[tid] = newadj[(size_t)b * 196 + tid];
    __syncthreads();
    const float* xb = x + (size_t)b * KN * C_DIM;
    uint32_t* mb = mbf32 + (size_t)b * KN * (C_DIM / 2);
    for (int c = tid * 2; c < C_DIM; c += 512) {
        float xl[KN], xh[KN];
        #pragma unroll
        for (int j = 0; j < KN; ++j) {
            xl[j] = xb[j * C_DIM + c];
            xh[j] = xb[j * C_DIM + c + 1];
        }
        #pragma unroll
        for (int i = 0; i < KN; ++i) {
            float al = 0.f, ah = 0.f;
            #pragma unroll
            for (int j = 0; j < KN; ++j) {
                float w = A[i * KN + j];
                al += w * xl[j];
                ah += w * xh[j];
            }
            mb[(i * C_DIM + c) >> 1] = pack_bf16(al, ah);
        }
    }
}

// ---------------- fused dual GEMM: out = relu(merged @ Wm) + x @ Wo ----------------
// 256 threads = 8 waves; tile 112x128; wave w owns N-subtile w, 7 M-subtiles.
__global__ __launch_bounds__(256)
void gemm_out_kernel(const float* __restrict__ x, const uint32_t* __restrict__ mbf32,
                     const float* __restrict__ wm, const float* __restrict__ wo,
                     float* __restrict__ out) {
    __shared__ __attribute__((aligned(16))) __bf16 a_bf[MB * KT];   // [row][k]
    __shared__ __attribute__((aligned(16))) __bf16 w_t[NB * KT];    // [n][k] (transposed)

    const int tid = threadIdx.x;
    const int wave = tid >> 5;
    const int lane = tid & 31;
    const int halfsel = lane >> 4;       // 0: K in {0..7,16..23}; 1: K in {8..15,24..31}
    const int kb = halfsel * 8;
    const int lm = lane & 15;
    const int nbase = blockIdx.x * NB;   // N fastest -> weights stay L2 resident
    const int mbase = blockIdx.y * MB;

    uint32_t* a32 = (uint32_t*)a_bf;
    uint32_t* w32 = (uint32_t*)w_t;

    v8f acc0[7], acc1[7];
    const v8f vzero = {0.f, 0.f, 0.f, 0.f, 0.f, 0.f, 0.f, 0.f};
    #pragma unroll
    for (int t = 0; t < 7; ++t) { acc0[t] = vzero; acc1[t] = vzero; }

    const int bcol = (wave * 16 + lm) * KT;     // this wave's B column row in w_t

    // =================== phase 0: A = merged(bf16), W = wm ===================
    for (int k0 = 0; k0 < C_DIM; k0 += KT) {
        __syncthreads();
        // A tile: straight dword copy, 112*32 dwords
        #pragma unroll
        for (int r = 0; r < 14; ++r) {
            int d = tid + r * 256;
            int row = d >> 5, kk2 = d & 31;
            a32[row * 32 + kk2] =
                mbf32[(size_t)(mbase + row) * (C_DIM / 2) + (k0 >> 1) + kk2];
        }
        // W tile: pack K-pairs, transposed layout [n][k]
        #pragma unroll
        for (int r = 0; r < 16; ++r) {
            int d = tid + r * 256;
            int kk2 = d >> 7, nn = d & 127;     // nn fastest -> coalesced
            const float* wp = &wm[(size_t)(k0 + 2 * kk2) * C_DIM + nbase + nn];
            w32[nn * 32 + kk2] = pack_bf16(wp[0], wp[C_DIM]);
        }
        __syncthreads();
        #pragma unroll
        for (int h = 0; h < 2; ++h) {
            union { v16bf v; v8bf p[2]; } bu;
            const int bo = bcol + h * 32 + kb;
            bu.p[0] = *(const v8bf*)&w_t[bo];
            bu.p[1] = *(const v8bf*)&w_t[bo + 16];
            #pragma unroll
            for (int t = 0; t < 7; ++t) {
                union { v16bf v; v8bf p[2]; } au;
                const int ao = (t * 16 + lm) * KT + h * 32 + kb;
                au.p[0] = *(const v8bf*)&a_bf[ao];
                au.p[1] = *(const v8bf*)&a_bf[ao + 16];
                acc0[t] = __builtin_amdgcn_wmma_f32_16x16x32_bf16(
                    false, au.v, false, bu.v, (short)0, acc0[t], false, false);
            }
        }
    }
    // =================== phase 1: A = x(f32->bf16), W = wo ===================
    for (int k0 = 0; k0 < C_DIM; k0 += KT) {
        __syncthreads();
        #pragma unroll
        for (int r = 0; r < 14; ++r) {
            int d = tid + r * 256;
            int row = d >> 5, kk2 = d & 31;
            const float* xp = &x[(size_t)(mbase + row) * C_DIM + k0 + 2 * kk2];
            a32[row * 32 + kk2] = pack_bf16(xp[0], xp[1]);
        }
        #pragma unroll
        for (int r = 0; r < 16; ++r) {
            int d = tid + r * 256;
            int kk2 = d >> 7, nn = d & 127;
            const float* wp = &wo[(size_t)(k0 + 2 * kk2) * C_DIM + nbase + nn];
            w32[nn * 32 + kk2] = pack_bf16(wp[0], wp[C_DIM]);
        }
        __syncthreads();
        #pragma unroll
        for (int h = 0; h < 2; ++h) {
            union { v16bf v; v8bf p[2]; } bu;
            const int bo = bcol + h * 32 + kb;
            bu.p[0] = *(const v8bf*)&w_t[bo];
            bu.p[1] = *(const v8bf*)&w_t[bo + 16];
            #pragma unroll
            for (int t = 0; t < 7; ++t) {
                union { v16bf v; v8bf p[2]; } au;
                const int ao = (t * 16 + lm) * KT + h * 32 + kb;
                au.p[0] = *(const v8bf*)&a_bf[ao];
                au.p[1] = *(const v8bf*)&a_bf[ao + 16];
                acc1[t] = __builtin_amdgcn_wmma_f32_16x16x32_bf16(
                    false, au.v, false, bu.v, (short)0, acc1[t], false, false);
            }
        }
    }
    // D layout: VGPR v, lanes0-15 -> M=v, lanes16-31 -> M=v+8; N = lane%16
    #pragma unroll
    for (int t = 0; t < 7; ++t) {
        #pragma unroll
        for (int v = 0; v < 8; ++v) {
            int m = mbase + t * 16 + v + 8 * halfsel;
            int n = nbase + wave * 16 + lm;
            float r0 = acc0[t][v];
            r0 = r0 > 0.f ? r0 : 0.f;
            out[(size_t)m * C_DIM + n] = r0 + acc1[t][v];
        }
    }
}

extern "C" void kernel_launch(void* const* d_in, const int* in_sizes, int n_in,
                              void* d_out, int out_size, void* d_ws, size_t ws_size,
                              hipStream_t stream) {
    const float* x     = (const float*)d_in[0];
    const float* adjm  = (const float*)d_in[1];
    const int*   ei    = (const int*)d_in[2];
    const int*   ej    = (const int*)d_in[3];
    const float* wdir  = (const float*)d_in[4];
    const float* gamma = (const float*)d_in[5];
    const float* beta  = (const float*)d_in[6];
    const float* wm    = (const float*)d_in[7];
    const float* wo    = (const float*)d_in[8];
    float* out = (float*)d_out;
    const int E  = in_sizes[2];
    const int bs = in_sizes[0] / (KN * C_DIM);

    float* ws      = (float*)d_ws;
    float* Asum    = ws;               // [2048]
    float* Bsum    = ws + 2048;        // [2048]
    float* coef    = ws + 4096;        // [2048]
    float* bias    = ws + 6144;        // [1]
    float* newadj  = ws + 8192;        // [bs*196]
    uint32_t* mbf32 = (uint32_t*)(ws + 8192 + (size_t)bs * 196);  // [bs*14*1024] dwords of bf16x2

    init_kernel<<<32, 256, 0, stream>>>(ws);
    stats_kernel<<<bs, 256, 0, stream>>>(x, ei, ej, E, Asum, Bsum);
    coef_kernel<<<1, 1024, 0, stream>>>(Asum, Bsum, gamma, beta, wdir, bs, E, coef, bias);
    adj_kernel<<<bs, 256, 0, stream>>>(x, adjm, ei, ej, E, coef, bias, newadj);
    merged_kernel<<<bs, 256, 0, stream>>>(x, newadj, mbf32);
    dim3 grid(C_DIM / NB, (bs * KN) / MB);   // N fastest for L2 weight residency
    gemm_out_kernel<<<grid, 256, 0, stream>>>(x, mbf32, wm, wo, out);
}
